// ConvictionPlanner_16020228014626
// MI455X (gfx1250) — matrix operand order repaired and checked by reading, not compiled, not run on hardware
//
#include <hip/hip_runtime.h>
#include <hip/hip_bf16.h>
#include <math.h>

typedef __bf16 bf16_t;
typedef bf16_t v16bf __attribute__((ext_vector_type(16)));
typedef bf16_t v8bf  __attribute__((ext_vector_type(8)));
typedef float  v8f   __attribute__((ext_vector_type(8)));

#define D_DIM   512
#define N_KEYS  16384
#define B_ROWS  1024
#define A_ACT   5
#define TOPK    50
#define NSTEPS  5
#define NTILES  69            // ceil((1024 + 5*15)/16)
#define PERM_STRIDE (NTILES*16)   // 1104

// ---------------------------------------------------------------------------
// Row L2-normalize (f32 -> bf16). One wave per row, 8 rows per 256-thread WG.
// ---------------------------------------------------------------------------
__global__ __launch_bounds__(256)
void normalize_rows_kernel(const float* __restrict__ src, bf16_t* __restrict__ dst, int nrows)
{
    int row  = blockIdx.x * 8 + (threadIdx.x >> 5);
    int lane = threadIdx.x & 31;
    if (row >= nrows) return;
    const float4* s4 = (const float4*)(src + (size_t)row * D_DIM);
    float4 v[4];
    float ss = 0.f;
    #pragma unroll
    for (int i = 0; i < 4; ++i) {
        v[i] = s4[lane + i * 32];
        ss += v[i].x*v[i].x + v[i].y*v[i].y + v[i].z*v[i].z + v[i].w*v[i].w;
    }
    #pragma unroll
    for (int off = 16; off; off >>= 1) ss += __shfl_xor(ss, off, 32);
    float inv = 1.0f / (sqrtf(ss) + 1e-8f);
    bf16_t* d = dst + (size_t)row * D_DIM;
    #pragma unroll
    for (int i = 0; i < 4; ++i) {
        int base = (lane + i * 32) * 4;
        d[base + 0] = (bf16_t)(v[i].x * inv);
        d[base + 1] = (bf16_t)(v[i].y * inv);
        d[base + 2] = (bf16_t)(v[i].z * inv);
        d[base + 3] = (bf16_t)(v[i].w * inv);
    }
}

// ---------------------------------------------------------------------------
// Bucket rows by action per step; pad each bucket to multiple of 16 with -1.
// ---------------------------------------------------------------------------
__global__ __launch_bounds__(256)
void build_perms_kernel(const int* __restrict__ actions,
                        int* __restrict__ perm, int* __restrict__ tile_act)
{
    __shared__ int cnt[A_ACT], off[A_ACT], cur[A_ACT];
    for (int t = 0; t < NSTEPS; ++t) {
        int* permt = perm + t * PERM_STRIDE;
        for (int i = threadIdx.x; i < PERM_STRIDE; i += 256) permt[i] = -1;
        if (threadIdx.x < A_ACT) { cnt[threadIdx.x] = 0; cur[threadIdx.x] = 0; }
        __syncthreads();
        for (int b = threadIdx.x; b < B_ROWS; b += 256)
            atomicAdd(&cnt[actions[b * NSTEPS + t]], 1);
        __syncthreads();
        if (threadIdx.x == 0) {
            int o = 0;
            int* ta = tile_act + t * NTILES;
            for (int a = 0; a < A_ACT; ++a) {
                off[a] = o;
                int padded = (cnt[a] + 15) & ~15;
                for (int tt = o / 16; tt < (o + padded) / 16; ++tt) ta[tt] = a;
                o += padded;
            }
            for (int tt = o / 16; tt < NTILES; ++tt) ta[tt] = 0;
        }
        __syncthreads();
        for (int b = threadIdx.x; b < B_ROWS; b += 256) {
            int a = actions[b * NSTEPS + t];
            int pos = off[a] + atomicAdd(&cur[a], 1);
            permt[pos] = b;
        }
        __syncthreads();
    }
}

// ---------------------------------------------------------------------------
// sims[b][n] = qn[b] . kn[act(tile)][n], bf16 in / f32 out via WMMA.
// Wave computes a 16x16 C tile, K=512 fully unrolled -> 16 v_wmma per wave.
// Grid: (N_KEYS/128, NTILES), block 256 (8 waves -> 128 keys per WG).
// ---------------------------------------------------------------------------
__global__ __launch_bounds__(256)
void gemm_sims_kernel(const bf16_t* __restrict__ qnb,
                      const bf16_t* __restrict__ knb,
                      const int* __restrict__ perm,      // this step's 1104 rows
                      const int* __restrict__ tile_act,  // this step's 69 tile actions
                      float* __restrict__ sims)
{
    const int lane  = threadIdx.x & 31;
    const int wave  = threadIdx.x >> 5;
    const int rt    = blockIdx.y;
    const int nbase = blockIdx.x * 128 + wave * 16;
    const int act   = tile_act[rt];
    const int rsub  = lane & 15;
    const int khalf = lane >> 4;

    int brow = perm[rt * 16 + rsub];
    const bf16_t* arow = qnb + (size_t)(brow < 0 ? 0 : brow) * D_DIM;
    const bf16_t* brw  = knb + ((size_t)act * N_KEYS + (nbase + rsub)) * D_DIM;

    v8f acc = {};
    #pragma unroll
    for (int k0 = 0; k0 < D_DIM; k0 += 32) {
        // A (16x32, MxK): lane holds row rsub; K chunks [khalf*8,+8) and [16+khalf*8,+8)
        v8bf alo = *(const v8bf*)(arow + k0 + khalf * 8);
        v8bf ahi = *(const v8bf*)(arow + k0 + 16 + khalf * 8);
        // B (32x16, KxN): lane holds column rsub; K chunks [khalf*8,+8) and [16+khalf*8,+8)
        v8bf blo = *(const v8bf*)(brw + k0 + khalf * 8);
        v8bf bhi = *(const v8bf*)(brw + k0 + 16 + khalf * 8);
        v16bf A, Bm;
        #pragma unroll
        for (int i = 0; i < 8; ++i) {
            A[i] = alo[i];  A[8 + i] = ahi[i];
            Bm[i] = blo[i]; Bm[8 + i] = bhi[i];
        }
        acc = __builtin_amdgcn_wmma_f32_16x16x32_bf16(false, A, false, Bm,
                                                      (short)0, acc, false, false);
    }
    // C 16x16 f32: VGPR v -> row v + 8*khalf, col = lane&15
    const int col = nbase + rsub;
    #pragma unroll
    for (int v = 0; v < 8; ++v) {
        int b = perm[rt * 16 + v + 8 * khalf];
        if (b >= 0) sims[(size_t)b * N_KEYS + col] = acc[v];
    }
}

// ---------------------------------------------------------------------------
// Per-row: top-50 (iterative parallel argmax over LDS copy), softmax,
// weighted gather of mem_values, write cur (f32) + normalized bf16 query.
// One 256-thread WG per row; 64KB dynamic LDS for the sims row.
// ---------------------------------------------------------------------------
__global__ __launch_bounds__(256)
void select_kernel(const float* __restrict__ sims,
                   const float* __restrict__ mem_values,
                   const int*  __restrict__ actions,
                   int t,
                   float* __restrict__ curout,     // [1024][512] for this step
                   bf16_t* __restrict__ qnb)       // next-step normalized queries
{
    extern __shared__ float sv[];                  // 16384 floats (64 KB)
    __shared__ float wmaxv[8]; __shared__ int wmaxi[8];
    __shared__ float selv[TOPK]; __shared__ int seli[TOPK];
    __shared__ float sexp[TOPK];
    __shared__ float red[8];
    __shared__ float sscal;

    const int b    = blockIdx.x;
    const int tid  = threadIdx.x;
    const int lane = tid & 31, wave = tid >> 5;
    const int a    = actions[b * NSTEPS + t];

    const float4* srow = (const float4*)(sims + (size_t)b * N_KEYS);
    for (int i = tid; i < N_KEYS / 4; i += 256) ((float4*)sv)[i] = srow[i];
    __syncthreads();

    for (int k = 0; k < TOPK; ++k) {
        float m = -3.0e38f; int mi = 0x7fffffff;
        for (int i = tid; i < N_KEYS; i += 256) {
            float v = sv[i];
            if (v > m) { m = v; mi = i; }          // first hit keeps lowest index
        }
        #pragma unroll
        for (int off = 16; off; off >>= 1) {
            float om = __shfl_xor(m, off, 32);
            int   oi = __shfl_xor(mi, off, 32);
            if (om > m || (om == m && oi < mi)) { m = om; mi = oi; }
        }
        if (lane == 0) { wmaxv[wave] = m; wmaxi[wave] = mi; }
        __syncthreads();
        if (tid == 0) {
            float bm = wmaxv[0]; int bi = wmaxi[0];
            for (int w = 1; w < 8; ++w)
                if (wmaxv[w] > bm || (wmaxv[w] == bm && wmaxi[w] < bi)) { bm = wmaxv[w]; bi = wmaxi[w]; }
            selv[k] = bm; seli[k] = bi;
            sv[bi] = -3.4e38f;                     // knock out for next round
        }
        __syncthreads();
    }

    // softmax over the 50 (selv[0] is the max since selection is descending)
    if (tid < TOPK) sexp[tid] = expf(selv[tid] - selv[0]);
    __syncthreads();
    if (tid == 0) { float s = 0.f; for (int k = 0; k < TOPK; ++k) s += sexp[k]; sscal = 1.0f / s; }
    __syncthreads();
    const float invs = sscal;

    // weighted sum of gathered mem_values rows; thread owns d = tid, tid+256
    const float* mv = mem_values + (size_t)a * N_KEYS * D_DIM;
    float acc0 = 0.f, acc1 = 0.f;
    const int d0 = tid, d1 = tid + 256;
    for (int k = 0; k < TOPK; ++k) {
        const float* row = mv + (size_t)seli[k] * D_DIM;
        float w = sexp[k] * invs;
        acc0 += w * row[d0];
        acc1 += w * row[d1];
    }
    curout[(size_t)b * D_DIM + d0] = acc0;
    curout[(size_t)b * D_DIM + d1] = acc1;

    // L2-normalize the new query, emit bf16 for next step's GEMM
    float ss = acc0 * acc0 + acc1 * acc1;
    #pragma unroll
    for (int off = 16; off; off >>= 1) ss += __shfl_xor(ss, off, 32);
    if (lane == 0) red[wave] = ss;
    __syncthreads();
    if (tid == 0) { float s = 0.f; for (int w = 0; w < 8; ++w) s += red[w]; sscal = 1.0f / (sqrtf(s) + 1e-8f); }
    __syncthreads();
    const float inv = sscal;
    qnb[(size_t)b * D_DIM + d0] = (bf16_t)(acc0 * inv);
    qnb[(size_t)b * D_DIM + d1] = (bf16_t)(acc1 * inv);
}

// ---------------------------------------------------------------------------
// Fused MLP for all 6144 rows (1024 embeddings + 5*1024 step outputs).
// out index == row index (pred_nstep layout is b*5 + t).
// 8-row register blocking so W1/W2 are read once per 8 rows.
// ---------------------------------------------------------------------------
__global__ __launch_bounds__(256)
void mlp_kernel(const float* __restrict__ embeddings,
                const float* __restrict__ cur,      // [5][1024][512]
                const float* __restrict__ W1, const float* __restrict__ b1,
                const float* __restrict__ W2, const float* __restrict__ b2,
                const float* __restrict__ W3, const float* __restrict__ b3,
                float* __restrict__ out)
{
    __shared__ float xs[8][D_DIM];
    __shared__ float h1s[8][256];
    __shared__ float h2s[8][128];

    const int tid = threadIdx.x;
    const int rowbase = blockIdx.x * 64;

    for (int chunk = 0; chunk < 64; chunk += 8) {
        for (int r = 0; r < 8; ++r) {
            int row = rowbase + chunk + r;
            const float* src;
            if (row < B_ROWS) src = embeddings + (size_t)row * D_DIM;
            else {
                int q = row - B_ROWS; int bb = q / NSTEPS, tt = q % NSTEPS;
                src = cur + ((size_t)tt * B_ROWS + bb) * D_DIM;
            }
            for (int d = tid; d < D_DIM; d += 256) xs[r][d] = src[d];
        }
        __syncthreads();

        {   // layer 1: 512 -> 256, ELU
            float acc[8];
            #pragma unroll
            for (int r = 0; r < 8; ++r) acc[r] = b1[tid];
            for (int d = 0; d < D_DIM; ++d) {
                float w = W1[d * 256 + tid];
                #pragma unroll
                for (int r = 0; r < 8; ++r) acc[r] += xs[r][d] * w;
            }
            #pragma unroll
            for (int r = 0; r < 8; ++r) {
                float v = acc[r];
                h1s[r][tid] = v > 0.f ? v : (expf(v) - 1.f);
            }
        }
        __syncthreads();

        if (tid < 128) {   // layer 2: 256 -> 128, ELU
            float acc[8];
            #pragma unroll
            for (int r = 0; r < 8; ++r) acc[r] = b2[tid];
            for (int c = 0; c < 256; ++c) {
                float w = W2[c * 128 + tid];
                #pragma unroll
                for (int r = 0; r < 8; ++r) acc[r] += h1s[r][c] * w;
            }
            #pragma unroll
            for (int r = 0; r < 8; ++r) {
                float v = acc[r];
                h2s[r][tid] = v > 0.f ? v : (expf(v) - 1.f);
            }
        }
        __syncthreads();

        {   // layer 3: 128 -> 1; wave w handles row w
            int w = tid >> 5, l = tid & 31;
            float s = h2s[w][l]      * W3[l]      + h2s[w][l + 32] * W3[l + 32]
                    + h2s[w][l + 64] * W3[l + 64] + h2s[w][l + 96] * W3[l + 96];
            #pragma unroll
            for (int off = 16; off; off >>= 1) s += __shfl_xor(s, off, 32);
            if (l == 0) out[rowbase + chunk + w] = s + b3[0];
        }
        __syncthreads();
    }
}

// ---------------------------------------------------------------------------
// Workspace layout (bytes, 256-aligned)
// ---------------------------------------------------------------------------
static constexpr size_t OFF_KNB  = 0;                                   // 5*16384*512*2 = 83,886,080
static constexpr size_t OFF_QNB  = 83886080;                            // 1024*512*2    =  1,048,576
static constexpr size_t OFF_SIMS = OFF_QNB + 1048576;                   // 1024*16384*4  = 67,108,864
static constexpr size_t OFF_CUR  = OFF_SIMS + 67108864;                 // 5*1024*512*4  = 10,485,760
static constexpr size_t OFF_PERM = OFF_CUR + 10485760;                  // 5*1104*4      =     22,080
static constexpr size_t OFF_TACT = OFF_PERM + 22528;                    // 5*69*4        =      1,380

extern "C" void kernel_launch(void* const* d_in, const int* in_sizes, int n_in,
                              void* d_out, int out_size, void* d_ws, size_t ws_size,
                              hipStream_t stream)
{
    const float* embeddings = (const float*)d_in[0];
    const float* keys       = (const float*)d_in[1];
    const float* mem_values = (const float*)d_in[2];
    const float* W1 = (const float*)d_in[3];
    const float* b1 = (const float*)d_in[4];
    const float* W2 = (const float*)d_in[5];
    const float* b2 = (const float*)d_in[6];
    const float* W3 = (const float*)d_in[7];
    const float* b3 = (const float*)d_in[8];
    const int*   actions = (const int*)d_in[9];

    char* ws = (char*)d_ws;
    bf16_t* knb   = (bf16_t*)(ws + OFF_KNB);
    bf16_t* qnb   = (bf16_t*)(ws + OFF_QNB);
    float*  sims  = (float*) (ws + OFF_SIMS);
    float*  cur   = (float*) (ws + OFF_CUR);
    int*    perm  = (int*)   (ws + OFF_PERM);
    int*    tact  = (int*)   (ws + OFF_TACT);

    // Hoisted: normalize all keys to bf16 (84 MB, L2-resident thereafter)
    normalize_rows_kernel<<<(A_ACT * N_KEYS) / 8, 256, 0, stream>>>(keys, knb, A_ACT * N_KEYS);
    // Step-0 queries = normalized embeddings
    normalize_rows_kernel<<<B_ROWS / 8, 256, 0, stream>>>(embeddings, qnb, B_ROWS);
    // Action buckets for all 5 steps (actions known upfront)
    build_perms_kernel<<<1, 256, 0, stream>>>(actions, perm, tact);

    for (int t = 0; t < NSTEPS; ++t) {
        gemm_sims_kernel<<<dim3(N_KEYS / 128, NTILES), 256, 0, stream>>>(
            qnb, knb, perm + t * PERM_STRIDE, tact + t * NTILES, sims);
        select_kernel<<<B_ROWS, 256, N_KEYS * sizeof(float), stream>>>(
            sims, mem_values, actions, t, cur + (size_t)t * B_ROWS * D_DIM, qnb);
    }

    mlp_kernel<<<(B_ROWS * 6) / 64, 256, 0, stream>>>(
        embeddings, cur, W1, b1, W2, b2, W3, b3, (float*)d_out);
}
